// encoder_36129264894163
// MI455X (gfx1250) — compile-verified, matrix operand order checked by reading, not run on hardware
//
#include <hip/hip_runtime.h>

typedef unsigned short u16;
typedef unsigned int   u32;
typedef __attribute__((ext_vector_type(16))) __bf16 v16bf;
typedef __attribute__((ext_vector_type(8)))  float  v8f;

#define Bc   8
#define Tt   20
#define Hc   64
#define Wc   64
#define Cc   32
#define Fc   64
#define NG   256

// LDS channel strides (elements), padded; *2 bytes stays 16B-aligned per pixel
#define XSS  40
#define HSS  72

#define HS_ELEMS    ((size_t)Bc*Tt*Hc*Wc*Fc)       // 41,943,040
#define STATE_ELEMS ((size_t)Bc*Hc*Wc*Fc)          // 2,097,152
#define WPACK_ELEMS (27*8192)                      // 221,184 bf16
#define WS_CBUF_OFF 458752                         // bytes, after packed weights (padded)

__device__ __forceinline__ u16 f2bf(float f) {
    u32 u = __float_as_uint(f);
    u32 r = u + 0x7FFFu + ((u >> 16) & 1u);
    return (u16)(r >> 16);
}

__device__ __forceinline__ float hsig(float z) {
    return fminf(fmaxf(0.2f * z + 0.5f, 0.0f), 1.0f);
}

// ---------------------------------------------------------------------------
// Pack Wx [3,3,32,256] and Wh [3,3,64,256] (f32, HWIO) into bf16 B-fragments.
// Layout: [chunk 0..26][nsub 0..15][lane 0..31][e 0..15] contiguous bf16.
// chunk 0..8  : x taps (ky*3+kx), K = 32 input channels
// chunk 9..26 : h taps, 2 chunks/tap (part = K-half of the 64 h channels)
// Per-lane element->K mapping follows CDNA5 16-bit B layout:
//   n = nsub*16 + (lane&15); hi = lane>>4; k = e<8 ? hi*8+e : 16+hi*8+(e-8)
// ---------------------------------------------------------------------------
__global__ void pack_weights_kernel(const float* __restrict__ Wx,
                                    const float* __restrict__ Wh,
                                    u16* __restrict__ wp) {
    int idx = blockIdx.x * 256 + threadIdx.x;
    if (idx >= WPACK_ELEMS) return;
    int chunk = idx >> 13;
    int rem   = idx & 8191;
    int nsub  = rem >> 9;
    int lane  = (rem >> 4) & 31;
    int e     = rem & 15;
    int n     = nsub * 16 + (lane & 15);
    int hi    = lane >> 4;
    int k     = (e < 8) ? (hi * 8 + e) : (16 + hi * 8 + (e - 8));
    float v;
    if (chunk < 9) {
        v = Wx[((size_t)chunk * Cc + k) * NG + n];
    } else {
        int c2 = chunk - 9;
        int pos = c2 >> 1, part = c2 & 1;
        v = Wh[((size_t)pos * Fc + part * 32 + k) * NG + n];
    }
    wp[idx] = f2bf(v);
}

// ---------------------------------------------------------------------------
// One ConvLSTM timestep. Block = one (b, y) image row: 64 pixels x 256 gates.
// 256 threads = 8 waves. Wave w: a = w>>2 selects M half (2 of 4 M-subtiles),
// sf = w&3 selects f-channel subtile; gates handled via nsub = gate*4 + sf so
// each lane owns z_i/z_f/z_c/z_o for the same (pixel, f-channel).
// All 27 K-chunks fully unrolled: per-chunk tap/offset constants fold into
// ds_load / global_load immediate offsets.
// ---------------------------------------------------------------------------
__global__ __launch_bounds__(256)
void convlstm_step_kernel(const float* __restrict__ x,
                          const float* __restrict__ bias,
                          const u16* __restrict__ wp,
                          float* __restrict__ out,   // [B,T,H,W,F], t-1 read / t write
                          float* __restrict__ cbuf,  // [B,H,W,F]
                          int t,
                          float* __restrict__ hT,    // out tail or null
                          float* __restrict__ cT) {  // out tail or null
    __shared__ u16 xs [3 * 66 * XSS];   // bf16, cols padded (+1 each side, zero)
    __shared__ u16 hsm[3 * 66 * HSS];

    const int tid = threadIdx.x;
    const int bi  = blockIdx.x >> 6;
    const int y   = blockIdx.x & 63;

    // ---- zero-fill LDS (handles SAME padding + t==0 h state) ----
    for (int i = tid; i < (3 * 66 * XSS) / 2; i += 256) ((u32*)xs)[i]  = 0u;
    for (int i = tid; i < (3 * 66 * HSS) / 2; i += 256) ((u32*)hsm)[i] = 0u;
    __syncthreads();

    // ---- stage x rows y-1..y+1 : 3*64*32 f32 -> bf16 ----
    {
        const float* xbase = x + ((size_t)(bi * Tt + t) * (Hc * Wc)) * Cc;
        for (int i = tid; i < 1536; i += 256) {         // 1536 float4
            int r   = i / 512;
            int rem = i - r * 512;
            int px  = rem >> 3;
            int c4  = (rem & 7) * 4;
            int gy  = y + r - 1;
            if ((unsigned)gy < Hc) {
                float4 v = *(const float4*)(xbase + ((size_t)gy * Wc + px) * Cc + c4);
                u32 p0 = (u32)f2bf(v.x) | ((u32)f2bf(v.y) << 16);
                u32 p1 = (u32)f2bf(v.z) | ((u32)f2bf(v.w) << 16);
                u32* dst = (u32*)&xs[(r * 66 + px + 1) * XSS + c4];
                dst[0] = p0; dst[1] = p1;
            }
        }
    }
    // ---- stage h_{t-1} rows y-1..y+1 : 3*64*64 f32 -> bf16 ----
    if (t > 0) {
        const float* hbase = out + ((size_t)(bi * Tt + (t - 1)) * (Hc * Wc)) * Fc;
        for (int i = tid; i < 3072; i += 256) {         // 3072 float4
            int r   = i >> 10;
            int rem = i & 1023;
            int px  = rem >> 4;
            int c4  = (rem & 15) * 4;
            int gy  = y + r - 1;
            if ((unsigned)gy < Hc) {
                float4 v = *(const float4*)(hbase + ((size_t)gy * Wc + px) * Fc + c4);
                u32 p0 = (u32)f2bf(v.x) | ((u32)f2bf(v.y) << 16);
                u32 p1 = (u32)f2bf(v.z) | ((u32)f2bf(v.w) << 16);
                u32* dst = (u32*)&hsm[(r * 66 + px + 1) * HSS + c4];
                dst[0] = p0; dst[1] = p1;
            }
        }
    }
    __syncthreads();

    const int w    = tid >> 5;
    const int lane = tid & 31;
    const int a    = w >> 2;          // M half: ms in {2a, 2a+1}
    const int sf   = w & 3;           // f-channel subtile
    const int m    = lane & 15;
    const int hi   = lane >> 4;

    v8f acc[2][4] = {};               // [m-subtile][gate]

    // Variable address parts, computed once; all per-chunk deltas are literals.
    const int pixBase = (2 * a) * 16 + m;                  // 0..63 (+kx later)
    const u16* xsBase  = &xs [pixBase * XSS + hi * 8];     // + (ky*66+kx)*XSS const
    const u16* hsBase  = &hsm[pixBase * HSS + hi * 8];     // + (ky*66+kx)*HSS + part*32
    const u16* wpBase  = wp + (size_t)sf * 512 + (size_t)lane * 16;  // + (chunk*16+g*4)*512

#define DO_WMMA(APTR_BASE, STRIDE, LDSOFF, CHUNK)                                   \
    do {                                                                            \
        union { v16bf v; uint4 q[2]; } A0u, A1u;                                    \
        const u16* p0_ = (APTR_BASE) + (LDSOFF);                                    \
        A0u.q[0] = *(const uint4*)(p0_);                                            \
        A0u.q[1] = *(const uint4*)(p0_ + 16);                                       \
        const u16* p1_ = p0_ + 16 * (STRIDE);                                       \
        A1u.q[0] = *(const uint4*)(p1_);                                            \
        A1u.q[1] = *(const uint4*)(p1_ + 16);                                       \
        _Pragma("unroll")                                                           \
        for (int g = 0; g < 4; ++g) {                                               \
            v16bf Bf = *(const v16bf*)(wpBase + ((CHUNK) * 16 + g * 4) * 512);      \
            acc[0][g] = __builtin_amdgcn_wmma_f32_16x16x32_bf16(                    \
                false, A0u.v, false, Bf, (short)0, acc[0][g], false, false);        \
            acc[1][g] = __builtin_amdgcn_wmma_f32_16x16x32_bf16(                    \
                false, A1u.v, false, Bf, (short)0, acc[1][g], false, false);        \
        }                                                                           \
    } while (0)

    // ---- 9 x-chunks (taps), K = 32 input channels each ----
#pragma unroll
    for (int pos = 0; pos < 9; ++pos) {
        const int ky = pos / 3, kx = pos % 3;              // compile-time
        DO_WMMA(xsBase, XSS, (ky * 66 + kx) * XSS, pos);
    }
    // ---- 18 h-chunks (taps x K-halves), K = 32 each ----
#pragma unroll
    for (int c2 = 0; c2 < 18; ++c2) {
        const int pos = c2 >> 1, part = c2 & 1;            // compile-time
        const int ky = pos / 3, kx = pos % 3;
        DO_WMMA(hsBase, HSS, (ky * 66 + kx) * HSS + part * 32, 9 + c2);
    }
#undef DO_WMMA

    // ---- gates + state update (f32) ----
    const int fch = sf * 16 + (lane & 15);
    const float b_i = bias[fch];
    const float b_f = bias[64 + fch];
    const float b_c = bias[128 + fch];
    const float b_o = bias[192 + fch];

    float* hout = out + ((size_t)(bi * Tt + t) * (Hc * Wc)) * Fc;

#pragma unroll
    for (int msI = 0; msI < 2; ++msI) {
        const int msBase = (2 * a + msI) * 16;
#pragma unroll
        for (int r = 0; r < 8; ++r) {
            int p = msBase + r + hi * 8;
            size_t sidx = (((size_t)bi * Hc + y) * Wc + p) * Fc + fch;   // state index
            float ig = hsig(acc[msI][0][r] + b_i);
            float fg = hsig(acc[msI][1][r] + b_f);
            float gg = tanhf(acc[msI][2][r] + b_c);
            float og = hsig(acc[msI][3][r] + b_o);
            float cold = cbuf[sidx];
            float cnew = fg * cold + ig * gg;
            float hnew = og * tanhf(cnew);
            cbuf[sidx] = cnew;
            hout[((size_t)y * Wc + p) * Fc + fch] = hnew;
            if (hT) { hT[sidx] = hnew; cT[sidx] = cnew; }
        }
    }
}

extern "C" void kernel_launch(void* const* d_in, const int* in_sizes, int n_in,
                              void* d_out, int out_size, void* d_ws, size_t ws_size,
                              hipStream_t stream) {
    const float* x  = (const float*)d_in[0];
    const float* Wx = (const float*)d_in[1];
    const float* Wh = (const float*)d_in[2];
    const float* b  = (const float*)d_in[3];
    float* out = (float*)d_out;

    u16*   wp   = (u16*)d_ws;
    float* cbuf = (float*)((char*)d_ws + WS_CBUF_OFF);

    hipMemsetAsync(cbuf, 0, STATE_ELEMS * sizeof(float), stream);
    pack_weights_kernel<<<(WPACK_ELEMS + 255) / 256, 256, 0, stream>>>(Wx, Wh, wp);

    float* hT_out = out + HS_ELEMS;                 // [B,H,W,F]
    float* cT_out = out + HS_ELEMS + STATE_ELEMS;   // [B,H,W,F]

    for (int t = 0; t < Tt; ++t) {
        float* hT = (t == Tt - 1) ? hT_out : nullptr;
        float* cT = (t == Tt - 1) ? cT_out : nullptr;
        convlstm_step_kernel<<<Bc * Hc, 256, 0, stream>>>(x, b, wp, out, cbuf, t, hT, cT);
    }
}
